// MRNN_6777458393411
// MI455X (gfx1250) — compile-verified, matrix operand order checked by reading, not run on hardware
//
#include <hip/hip_runtime.h>
#include <cmath>

// ---------------------------------------------------------------------------
// MRNN on gfx1250: bf16 WMMA (v_wmma_f32_16x16x32_bf16) everywhere, LDS-
// resident recurrent weights staged via GLOBAL_LOAD_ASYNC_TO_LDS_B128,
// persistent scan kernel with software grid barrier.
// Precision: bf16 storage / fp32 accumulate.
// ---------------------------------------------------------------------------

#define T_STEPS 512
#define BATCH   64
#define DIM_I   256
#define DIM_H   1024
#define DIM_K   1024
#define DIM_O   256
#define NWG_STEP 32            // persistent workgroups; each owns 32 N-columns
#define ROWP    (1024 + 8)     // LDS weight row pitch (bf16 elems), bank-safe

typedef __bf16 bf16_t;
typedef __attribute__((ext_vector_type(16))) __bf16 v16bf;
typedef __attribute__((ext_vector_type(8)))  __bf16 v8bf;
typedef __attribute__((ext_vector_type(8)))  float  v8f;

// ---- bf16 <-> f32 via bit ops (round-to-nearest-even), no native-cvt risk --
__device__ __forceinline__ bf16_t f2bf(float f) {
    union { float f; unsigned u; } x; x.f = f;
    unsigned r = (x.u + 0x7FFFu + ((x.u >> 16) & 1u)) >> 16;
    union { unsigned short s; bf16_t b; } y; y.s = (unsigned short)r;
    return y.b;
}
__device__ __forceinline__ float bf2f(bf16_t b) {
    union { unsigned short s; bf16_t b; } y; y.b = b;
    union { unsigned u; float f; } x; x.u = ((unsigned)y.s) << 16;
    return x.f;
}

// ---- 16-bit WMMA A/B fragment loader (ISA 7.12.2 layout) -------------------
// lane<16 : row = base+l16, K = [kb..kb+7] and [kb+16..kb+23]
// lane>=16: same row group, K = [kb+8..kb+15] and [kb+24..kb+31]
union FragU { v16bf v; v8bf h[2]; };
__device__ __forceinline__ v16bf load_frag(const bf16_t* __restrict__ base,
                                           int row_stride, int row, int kb, int hs) {
    FragU f;
    const bf16_t* p = base + (size_t)row * row_stride + kb + hs * 8;
    f.h[0] = *(const v8bf*)p;          // 16B load -> VGPR 0..3
    f.h[1] = *(const v8bf*)(p + 16);   // 16B load -> VGPR 4..7
    return f.v;
}

__device__ __forceinline__ v8f wmma_bf16(v16bf a, v16bf b, v8f c) {
    return __builtin_amdgcn_wmma_f32_16x16x32_bf16(false, a, false, b,
                                                   (short)0, c, false, false);
}

__device__ __forceinline__ v8f vzero8() {
    v8f z; for (int i = 0; i < 8; ++i) z[i] = 0.0f; return z;
}

// ---- CDNA5 async global->LDS copy (ASYNCcnt-tracked, no VGPR data) ---------
// LDS destination address = low 32 bits of the flat shared pointer
// (ISA 10.2: LDS aperture maps addr[31:0] to the wave-relative LDS offset).
__device__ __forceinline__ void async_g2l_b128(void* lds_ptr, const void* gptr) {
    unsigned lds_off = (unsigned)(size_t)lds_ptr;
    asm volatile("global_load_async_to_lds_b128 %0, %1, off"
                 :: "v"(lds_off), "v"(gptr) : "memory");
}
__device__ __forceinline__ void wait_async0() {
    asm volatile("s_wait_asynccnt 0" ::: "memory");
}

// ---- software grid barrier (L2 atomics, sense via generation counter) ------
__device__ __forceinline__ void grid_barrier(unsigned* cnt, unsigned* gen,
                                             unsigned nwg, unsigned& local_gen) {
    __syncthreads();
    __threadfence();
    if (threadIdx.x == 0) {
        unsigned prev = atomicAdd(cnt, 1u);
        if (prev == nwg - 1u) {
            atomicExch(cnt, 0u);
            atomicAdd(gen, 1u);
        } else {
            while (atomicAdd(gen, 0u) <= local_gen) {
                __builtin_amdgcn_s_sleep(2);
            }
        }
    }
    local_gen++;
    __syncthreads();
    __threadfence();
}

// ---------------------------------------------------------------------------
__global__ void mrnn_init_bar(unsigned* bar) { bar[0] = 0u; bar[1] = 0u; }

__global__ void mrnn_cvt_bf16(const float* __restrict__ src,
                              bf16_t* __restrict__ dst, int n) {
    int i = blockIdx.x * blockDim.x + threadIdx.x;
    if (i < n) dst[i] = f2bf(src[i]);
}

// ---- Precompute: [Xm | Xh] = X @ [W_mx | W_hx]^T  (M=32768, K=256) ---------
// grid (512, 16), 256 thr = 8 waves (2M x 4N); wave -> 32x32 (2x2 tiles);
// block = 64 x 128. 4 WMMAs per 8 b128 loads (2x the reuse of 1-tile waves).
__global__ void mrnn_gemm_pre(const bf16_t* __restrict__ Xbf,
                              const bf16_t* __restrict__ Wmx,
                              const bf16_t* __restrict__ Whx,
                              bf16_t* __restrict__ Xm,
                              bf16_t* __restrict__ Xh) {
    const int lane = threadIdx.x & 31, wave = threadIdx.x >> 5;
    const int l16 = lane & 15, hs = lane >> 4;
    const int wm = wave & 1, wn = wave >> 1;
    const int mbase = blockIdx.x * 64 + wm * 32;
    const int ncolbase = blockIdx.y * 128 + wn * 32;   // 1024 boundary is block-aligned

    const bf16_t* W; bf16_t* dst; int nw;
    if (ncolbase < DIM_K) { W = Wmx; dst = Xm; nw = ncolbase; }
    else                  { W = Whx; dst = Xh; nw = ncolbase - DIM_K; }

    v8f acc00 = vzero8(), acc01 = vzero8(), acc10 = vzero8(), acc11 = vzero8();
    for (int kb = 0; kb < DIM_I; kb += 32) {
        v16bf a0 = load_frag(Xbf, DIM_I, mbase + l16,      kb, hs);
        v16bf a1 = load_frag(Xbf, DIM_I, mbase + 16 + l16, kb, hs);
        v16bf b0 = load_frag(W,   DIM_I, nw + l16,         kb, hs);
        v16bf b1 = load_frag(W,   DIM_I, nw + 16 + l16,    kb, hs);
        acc00 = wmma_bf16(a0, b0, acc00);
        acc01 = wmma_bf16(a0, b1, acc01);
        acc10 = wmma_bf16(a1, b0, acc10);
        acc11 = wmma_bf16(a1, b1, acc11);
    }
    for (int r = 0; r < 8; ++r) {
        int r0 = mbase + r + 8 * hs, r1 = r0 + 16;
        int c0 = nw + l16,           c1 = c0 + 16;
        dst[(size_t)r0 * 1024 + c0] = f2bf(acc00[r]);
        dst[(size_t)r0 * 1024 + c1] = f2bf(acc01[r]);
        dst[(size_t)r1 * 1024 + c0] = f2bf(acc10[r]);
        dst[(size_t)r1 * 1024 + c1] = f2bf(acc11[r]);
    }
}

// ---- Persistent sequential scan --------------------------------------------
// 32 WGs x 256 thr. WG g owns output columns [g*32, g*32+32) of both GEMMs;
// its W_mh / W_hm slices (64KB each, bf16) live in LDS for all 512 steps,
// staged with GLOBAL_LOAD_ASYNC_TO_LDS_B128.
__global__ void mrnn_step(const bf16_t* __restrict__ Wmh,
                          const bf16_t* __restrict__ Whm,
                          const bf16_t* __restrict__ Xm,
                          const bf16_t* __restrict__ Xh,
                          const float*  __restrict__ b_h,
                          bf16_t* __restrict__ h_bf,
                          bf16_t* __restrict__ m_bf,
                          bf16_t* __restrict__ Hs,
                          float*  __restrict__ h_last,
                          unsigned* bar_cnt, unsigned* bar_gen) {
    extern __shared__ char smem[];
    bf16_t* wmh_s = (bf16_t*)smem;          // [32][ROWP]
    bf16_t* whm_s = wmh_s + 32 * ROWP;      // [32][ROWP]
    const int g = blockIdx.x;

    // stage both weight slices once (32 rows x 1024 cols, 16B async chunks)
    for (int c = threadIdx.x; c < 32 * 128; c += blockDim.x) {
        int row = c >> 7, col = (c & 127) << 3;
        async_g2l_b128(&wmh_s[row * ROWP + col],
                       &Wmh[(size_t)(g * 32 + row) * DIM_H + col]);
        async_g2l_b128(&whm_s[row * ROWP + col],
                       &Whm[(size_t)(g * 32 + row) * DIM_K + col]);
    }
    wait_async0();
    __syncthreads();

    const int lane = threadIdx.x & 31, wave = threadIdx.x >> 5;
    const int l16 = lane & 15, hs = lane >> 4;
    const int mt = wave & 3, nt = wave >> 2;       // 4 M-tiles x 2 N-tiles
    const int ncol = g * 32 + nt * 16 + l16;       // global output column
    unsigned local_gen = 0;

    for (int t = 0; t < T_STEPS; ++t) {
        const bf16_t* Xm_t = Xm + (size_t)t * BATCH * DIM_K;
        const bf16_t* Xh_t = Xh + (size_t)t * BATCH * DIM_H;

        // phase A: Hm = h @ W_mh^T ; m = Xm[t] * Hm   (K-dim = H)
        v8f acc = vzero8();
        for (int kb = 0; kb < DIM_H; kb += 32) {
            v16bf a = load_frag(h_bf,  DIM_H, mt * 16 + l16, kb, hs);
            v16bf b = load_frag(wmh_s, ROWP,  nt * 16 + l16, kb, hs);
            acc = wmma_bf16(a, b, acc);
        }
        for (int r = 0; r < 8; ++r) {
            int row = mt * 16 + r + 8 * hs;        // batch index 0..63
            float gate = bf2f(Xm_t[(size_t)row * DIM_K + ncol]);
            m_bf[(size_t)row * DIM_K + ncol] = f2bf(acc[r] * gate);
        }
        grid_barrier(bar_cnt, bar_gen, NWG_STEP, local_gen);

        // phase B: h = tanh(m @ W_hm^T + Xh[t] + b_h)   (K-dim = K)
        acc = vzero8();
        for (int kb = 0; kb < DIM_K; kb += 32) {
            v16bf a = load_frag(m_bf,  DIM_K, mt * 16 + l16, kb, hs);
            v16bf b = load_frag(whm_s, ROWP,  nt * 16 + l16, kb, hs);
            acc = wmma_bf16(a, b, acc);
        }
        float bias = b_h[ncol];
        for (int r = 0; r < 8; ++r) {
            int row = mt * 16 + r + 8 * hs;
            float v = acc[r] + bf2f(Xh_t[(size_t)row * DIM_H + ncol]) + bias;
            float hv = tanhf(v);
            bf16_t hb = f2bf(hv);
            h_bf[(size_t)row * DIM_H + ncol] = hb;
            Hs[(size_t)t * BATCH * DIM_H + (size_t)row * DIM_H + ncol] = hb;
            if (t == T_STEPS - 1) h_last[(size_t)row * DIM_H + ncol] = hv;
        }
        grid_barrier(bar_cnt, bar_gen, NWG_STEP, local_gen);
    }
}

// ---- Output GEMM: out = Hs @ W_oh^T + b_o  (M=32768, K=1024, N=256) --------
// grid (512, 2), block = 64 x 128, 2x2 tiles per wave. fp32 output.
__global__ void mrnn_gemm_out(const bf16_t* __restrict__ Hs,
                              const bf16_t* __restrict__ Woh,
                              const float*  __restrict__ b_o,
                              float* __restrict__ out) {
    const int lane = threadIdx.x & 31, wave = threadIdx.x >> 5;
    const int l16 = lane & 15, hs = lane >> 4;
    const int wm = wave & 1, wn = wave >> 1;
    const int mbase = blockIdx.x * 64 + wm * 32;
    const int nbase = blockIdx.y * 128 + wn * 32;

    v8f acc00 = vzero8(), acc01 = vzero8(), acc10 = vzero8(), acc11 = vzero8();
    for (int kb = 0; kb < DIM_H; kb += 32) {
        v16bf a0 = load_frag(Hs,  DIM_H, mbase + l16,      kb, hs);
        v16bf a1 = load_frag(Hs,  DIM_H, mbase + 16 + l16, kb, hs);
        v16bf b0 = load_frag(Woh, DIM_H, nbase + l16,      kb, hs);
        v16bf b1 = load_frag(Woh, DIM_H, nbase + 16 + l16, kb, hs);
        acc00 = wmma_bf16(a0, b0, acc00);
        acc01 = wmma_bf16(a0, b1, acc01);
        acc10 = wmma_bf16(a1, b0, acc10);
        acc11 = wmma_bf16(a1, b1, acc11);
    }
    float bias0 = b_o[nbase + l16], bias1 = b_o[nbase + 16 + l16];
    for (int r = 0; r < 8; ++r) {
        int r0 = mbase + r + 8 * hs, r1 = r0 + 16;
        int c0 = nbase + l16,        c1 = c0 + 16;
        out[(size_t)r0 * DIM_O + c0] = acc00[r] + bias0;
        out[(size_t)r0 * DIM_O + c1] = acc01[r] + bias1;
        out[(size_t)r1 * DIM_O + c0] = acc10[r] + bias0;
        out[(size_t)r1 * DIM_O + c1] = acc11[r] + bias1;
    }
}

// ---------------------------------------------------------------------------
extern "C" void kernel_launch(void* const* d_in, const int* in_sizes, int n_in,
                              void* d_out, int out_size, void* d_ws, size_t ws_size,
                              hipStream_t stream) {
    const float* x_f   = (const float*)d_in[0];   // [T,B,I]
    const float* h0_f  = (const float*)d_in[1];   // [B,H]
    const float* Wmx_f = (const float*)d_in[2];   // [K,I]
    const float* Wmh_f = (const float*)d_in[3];   // [K,H]
    const float* Whm_f = (const float*)d_in[4];   // [H,K]
    const float* Whx_f = (const float*)d_in[5];   // [H,I]
    const float* bh_f  = (const float*)d_in[6];   // [H]
    const float* Woh_f = (const float*)d_in[7];   // [O,H]
    const float* bo_f  = (const float*)d_in[8];   // [O]
    float* out = (float*)d_out;                   // [T,B,O] ++ [B,H]
    char*  ws  = (char*)d_ws;

    // workspace layout (all 256B-aligned)
    constexpr size_t OFF_BAR = 0;
    constexpr size_t OFF_WMX = 256;
    constexpr size_t OFF_WMH = OFF_WMX + (size_t)DIM_K * DIM_I * 2;
    constexpr size_t OFF_WHM = OFF_WMH + (size_t)DIM_K * DIM_H * 2;
    constexpr size_t OFF_WHX = OFF_WHM + (size_t)DIM_H * DIM_K * 2;
    constexpr size_t OFF_WOH = OFF_WHX + (size_t)DIM_H * DIM_I * 2;
    constexpr size_t OFF_XBF = OFF_WOH + (size_t)DIM_O * DIM_H * 2;
    constexpr size_t OFF_XM  = OFF_XBF + (size_t)T_STEPS * BATCH * DIM_I * 2;
    constexpr size_t OFF_XH  = OFF_XM  + (size_t)T_STEPS * BATCH * DIM_K * 2;
    constexpr size_t OFF_HBF = OFF_XH  + (size_t)T_STEPS * BATCH * DIM_H * 2;
    constexpr size_t OFF_MBF = OFF_HBF + (size_t)BATCH * DIM_H * 2;
    constexpr size_t OFF_HS  = OFF_MBF + (size_t)BATCH * DIM_K * 2;

    unsigned* bar  = (unsigned*)(ws + OFF_BAR);
    bf16_t* Wmx_b  = (bf16_t*)(ws + OFF_WMX);
    bf16_t* Wmh_b  = (bf16_t*)(ws + OFF_WMH);
    bf16_t* Whm_b  = (bf16_t*)(ws + OFF_WHM);
    bf16_t* Whx_b  = (bf16_t*)(ws + OFF_WHX);
    bf16_t* Woh_b  = (bf16_t*)(ws + OFF_WOH);
    bf16_t* Xbf    = (bf16_t*)(ws + OFF_XBF);
    bf16_t* Xm     = (bf16_t*)(ws + OFF_XM);
    bf16_t* Xh     = (bf16_t*)(ws + OFF_XH);
    bf16_t* Hbf    = (bf16_t*)(ws + OFF_HBF);
    bf16_t* Mbf    = (bf16_t*)(ws + OFF_MBF);
    bf16_t* Hs     = (bf16_t*)(ws + OFF_HS);

    mrnn_init_bar<<<1, 1, 0, stream>>>(bar);

    auto cvt = [&](const float* s, bf16_t* d, int n) {
        mrnn_cvt_bf16<<<(n + 255) / 256, 256, 0, stream>>>(s, d, n);
    };
    cvt(x_f,   Xbf,   T_STEPS * BATCH * DIM_I);
    cvt(h0_f,  Hbf,   BATCH * DIM_H);
    cvt(Wmx_f, Wmx_b, DIM_K * DIM_I);
    cvt(Wmh_f, Wmh_b, DIM_K * DIM_H);
    cvt(Whm_f, Whm_b, DIM_H * DIM_K);
    cvt(Whx_f, Whx_b, DIM_H * DIM_I);
    cvt(Woh_f, Woh_b, DIM_O * DIM_H);

    mrnn_gemm_pre<<<dim3(512, 16), 256, 0, stream>>>(Xbf, Wmx_b, Whx_b, Xm, Xh);

    size_t lds_bytes = 2u * 32u * ROWP * sizeof(bf16_t);  // ~129 KB of 320 KB
    mrnn_step<<<NWG_STEP, 256, lds_bytes, stream>>>(
        Wmh_b, Whm_b, Xm, Xh, bh_f, Hbf, Mbf, Hs,
        out + (size_t)T_STEPS * BATCH * DIM_O, bar, bar + 1);

    mrnn_gemm_out<<<dim3(512, 2), 256, 0, stream>>>(Hs, Woh_b, bo_f, out);
}
